// DGCNN_19146964205900
// MI455X (gfx1250) — matrix operand order repaired and checked by reading, not compiled, not run on hardware
//
#include <hip/hip_runtime.h>

// ---------------------------------------------------------------------------
// SGConv-style DGCNN forward, fused for MI455X (gfx1250, wave32, WMMA f16)
//   out[b] = relu( relu( M @ (x_b @ Wlin) + blin ) pooled by conv_w + cb ) @ Wfc + bfc
//   where M = A_norm^2 is batch-invariant and precomputed on device.
// ---------------------------------------------------------------------------

typedef _Float16 h4   __attribute__((ext_vector_type(4)));
typedef _Float16 h8   __attribute__((ext_vector_type(8)));
typedef _Float16 v16h __attribute__((ext_vector_type(16)));
typedef float    v8f  __attribute__((ext_vector_type(8)));

#define NN   128   // nodes
#define FIN  128   // input features
#define HH   64    // hiddens
#define CC   40    // classes
#define LP   136   // padded LDS row stride in halfs (272B -> conflict-free tiles)

// ------------------------- prep: M = A_norm^2, Wlin^T ----------------------
__global__ __launch_bounds__(256) void sgc_prep(
    const float* __restrict__ edge_w,   // [N*(N+1)/2] tril
    const float* __restrict__ lin_w,    // [FIN, HH]
    _Float16* __restrict__ Mg,          // out: [NN, NN] f16 row-major
    _Float16* __restrict__ WlT)         // out: [HH, FIN] f16 (= lin_w^T)
{
    __shared__ _Float16 As[NN * 132];
    __shared__ float    dinv[NN];
    const int tid = threadIdx.x;

    // degree of A_hat = A + I, then D^-1/2
    if (tid < NN) {
        const int i = tid;
        const int bi = (i * (i + 1)) >> 1;
        float s = 0.f;
        for (int j = 0; j < NN; ++j) {
            float a = (j <= i) ? edge_w[bi + j] : edge_w[((j * (j + 1)) >> 1) + i];
            if (j == i) a += 1.0f;
            s += a;
        }
        dinv[i] = (s > 0.f) ? rsqrtf(s) : 0.f;
    }
    __syncthreads();

    // A_norm = D^-1/2 (A + I) D^-1/2  (f16 in LDS)
    for (int idx = tid; idx < NN * NN; idx += 256) {
        const int i = idx >> 7, j = idx & 127;
        float a = (j <= i) ? edge_w[((i * (i + 1)) >> 1) + j]
                           : edge_w[((j * (j + 1)) >> 1) + i];
        if (j == i) a += 1.0f;
        As[i * 132 + j] = (_Float16)(a * dinv[i] * dinv[j]);
    }
    __syncthreads();

    // M = A_norm @ A_norm (K=2 hops collapsed), f32 accumulate -> f16
    for (int idx = tid; idx < NN * NN; idx += 256) {
        const int i = idx >> 7, j = idx & 127;
        float s = 0.f;
        for (int k = 0; k < NN; ++k)
            s += (float)As[i * 132 + k] * (float)As[k * 132 + j];
        Mg[idx] = (_Float16)s;
    }

    // WlT[h][f] = lin_w[f][h]
    for (int idx = tid; idx < HH * FIN; idx += 256) {
        const int h = idx >> 7, f = idx & 127;
        WlT[idx] = (_Float16)lin_w[f * HH + h];
    }
}

// ------------------------- main fused per-batch kernel ---------------------
__global__ __launch_bounds__(256) void sgc_main(
    const float*    __restrict__ x,      // [B, NN, FIN]
    const float*    __restrict__ lin_b,  // [HH]
    const float*    __restrict__ conv_w, // [NN]
    const float*    __restrict__ conv_b, // [1]
    const float*    __restrict__ fc_w,   // [HH, CC]
    const float*    __restrict__ fc_b,   // [CC]
    const _Float16* __restrict__ Mg,     // [NN, NN]
    const _Float16* __restrict__ WlT,    // [HH, FIN]
    float*          __restrict__ out)    // [B, CC]
{
    __shared__ _Float16 xs[NN * LP];   // x_b, f16, padded row-major
    __shared__ _Float16 ts[HH * LP];   // t = x_b @ Wlin, f16, column-major(hidden)
    __shared__ float    pooled[HH];

    const int tid  = threadIdx.x;
    const int b    = blockIdx.x;
    const int lane = tid & 31;
    const int w    = tid >> 5;            // wave id, 8 waves

    if (tid < HH) pooled[tid] = 0.f;

    // ---- stage x_b -> LDS (coalesced float4 stream, cvt to f16) ----
    const float* xb = x + (size_t)b * (NN * FIN);
    #pragma unroll
    for (int c = 0; c < 16; ++c) {
        const int flat = (c * 256 + tid) * 4;
        const int row  = flat >> 7, col = flat & 127;
        const float4 v = *(const float4*)(xb + flat);
        h4 hv = { (_Float16)v.x, (_Float16)v.y, (_Float16)v.z, (_Float16)v.w };
        *(h4*)&xs[row * LP + col] = hv;
    }
    __syncthreads();

    const int m    = lane & 15;           // row-in-tile (A) / col-in-tile (B)
    const int ks8  = (lane >> 4) << 3;    // unused; A-frag also uses ks16
    const int ks16 = (lane >> 4) << 4;    // B-frag K-chunk select
    const int rowb = w * 16 + ((lane >> 4) << 3);  // D-frag row base

    // ============ GEMM1: t = x_b @ Wlin  (rows block w) ============
    v16h afr[4];
    {
        const int ro = (w * 16 + m) * LP;
        #pragma unroll
        for (int k0 = 0; k0 < 4; ++k0) {
            h8 lo = *(const h8*)&xs[ro + k0 * 32 + ks16];
            h8 hi = *(const h8*)&xs[ro + k0 * 32 + ks16 + 8];
            afr[k0] = __builtin_shufflevector(lo, hi, 0,1,2,3,4,5,6,7,8,9,10,11,12,13,14,15);
        }
    }
    #pragma unroll
    for (int hc = 0; hc < 4; ++hc) {
        v8f acc = {};
        #pragma unroll
        for (int k0 = 0; k0 < 4; ++k0) {
            const _Float16* bp = WlT + (hc * 16 + m) * FIN + k0 * 32 + ks16;
            h8 lo = *(const h8*)bp;
            h8 hi = *(const h8*)(bp + 8);
            v16h bf = __builtin_shufflevector(lo, hi, 0,1,2,3,4,5,6,7,8,9,10,11,12,13,14,15);
            acc = __builtin_amdgcn_wmma_f32_16x16x32_f16(false, afr[k0], false, bf,
                                                         (short)0, acc, false, false);
        }
        // write t tile column-major(hidden): contiguous 8 halfs per lane
        h8 th;
        #pragma unroll
        for (int r = 0; r < 8; ++r) th[r] = (_Float16)acc[r];
        *(h8*)&ts[(hc * 16 + m) * LP + rowb] = th;
    }
    __syncthreads();

    // ============ GEMM2: h = M @ t  (rows block w) + fused epilogue ============
    {
        const _Float16* mr = Mg + (w * 16 + m) * NN;
        #pragma unroll
        for (int k0 = 0; k0 < 4; ++k0) {
            h8 lo = *(const h8*)(mr + k0 * 32 + ks16);
            h8 hi = *(const h8*)(mr + k0 * 32 + ks16 + 8);
            afr[k0] = __builtin_shufflevector(lo, hi, 0,1,2,3,4,5,6,7,8,9,10,11,12,13,14,15);
        }
    }
    #pragma unroll
    for (int hc = 0; hc < 4; ++hc) {
        v8f acc = {};
        #pragma unroll
        for (int k0 = 0; k0 < 4; ++k0) {
            const _Float16* bp = &ts[(hc * 16 + m) * LP + k0 * 32 + ks16];
            h8 lo = *(const h8*)bp;
            h8 hi = *(const h8*)(bp + 8);
            v16h bf = __builtin_shufflevector(lo, hi, 0,1,2,3,4,5,6,7,8,9,10,11,12,13,14,15);
            acc = __builtin_amdgcn_wmma_f32_16x16x32_f16(false, afr[k0], false, bf,
                                                         (short)0, acc, false, false);
        }
        // fused: +lin_b, relu, conv_w-weighted node pooling
        const int   col = hc * 16 + m;
        const float lb  = lin_b[col];
        float part = 0.f;
        #pragma unroll
        for (int r = 0; r < 8; ++r) {
            float v = acc[r] + lb;
            v = v > 0.f ? v : 0.f;
            part += conv_w[rowb + r] * v;
        }
        atomicAdd(&pooled[col], part);   // ds_add_f32
    }
    __syncthreads();

    // ---- final: relu(pooled + conv_b) @ fc_w + fc_b ----
    if (tid < CC) {
        const float cb = conv_b[0];
        float a = fc_b[tid];
        #pragma unroll 8
        for (int h = 0; h < HH; ++h) {
            float p = pooled[h] + cb;
            p = p > 0.f ? p : 0.f;
            a += p * fc_w[h * CC + tid];
        }
        out[(size_t)b * CC + tid] = a;
    }
}

// ---------------------------------------------------------------------------
extern "C" void kernel_launch(void* const* d_in, const int* in_sizes, int n_in,
                              void* d_out, int out_size, void* d_ws, size_t ws_size,
                              hipStream_t stream) {
    (void)n_in; (void)out_size; (void)ws_size;
    const float* x      = (const float*)d_in[0];
    const float* edge_w = (const float*)d_in[1];
    const float* lin_w  = (const float*)d_in[2];
    const float* lin_b  = (const float*)d_in[3];
    const float* conv_w = (const float*)d_in[4];
    const float* conv_b = (const float*)d_in[5];
    const float* fc_w   = (const float*)d_in[6];
    const float* fc_b   = (const float*)d_in[7];

    _Float16* Mg  = (_Float16*)d_ws;                          // 32 KB
    _Float16* WlT = (_Float16*)((char*)d_ws + NN * NN * 2);   // 16 KB

    const int B = in_sizes[0] / (NN * FIN);                    // 4096

    sgc_prep<<<1, 256, 0, stream>>>(edge_w, lin_w, Mg, WlT);
    sgc_main<<<B, 256, 0, stream>>>(x, lin_b, conv_w, conv_b, fc_w, fc_b,
                                    Mg, WlT, (float*)d_out);
}
